// AttentionPredictModel_40621800685586
// MI455X (gfx1250) — compile-verified, matrix operand order checked by reading, not run on hardware
//
#include <hip/hip_runtime.h>
#include <hip/hip_bf16.h>

typedef __attribute__((ext_vector_type(16))) _Float16 v16h;
typedef __attribute__((ext_vector_type(8)))  float    v8f;
typedef _Float16 half_t;

#define VOCAB 200000
#define Bsz   1024
#define Tlen  200
#define TClen 50
#define Ffeat 10
#define FKf   8
#define Krows 16          // 15 top-k + self
#define ROWS_PER_B 66     // 16 + 50
#define Mrows (Bsz*ROWS_PER_B)
#define Hdim  64
#define NEGV  (-1.0e9f)

// ---------------- wave32 reduction helpers ----------------
__device__ inline float wave_sum(float v){
  #pragma unroll
  for (int o = 16; o > 0; o >>= 1) v += __shfl_xor(v, o, 32);
  return v;
}
__device__ inline float wave_max(float v){
  #pragma unroll
  for (int o = 16; o > 0; o >>= 1) v = fmaxf(v, __shfl_xor(v, o, 32));
  return v;
}

// ---------------- WMMA fragment helpers (ISA 7.12.2 layouts) ----------------
// A: 16x32 f16 tile from row-major source, lda in halves.
// Per lane: two contiguous 8-half (16B) chunks -> 2x global_load_b128.
__device__ inline v16h load_frag_A(const half_t* A, int lda, int m0, int k0){
  int lane = threadIdx.x & 31;
  int m  = m0 + (lane & 15);
  int hi = lane >> 4;
  const half_t* base = A + (size_t)m * lda + k0 + hi * 8;
  v16h f;
  #pragma unroll
  for (int j = 0; j < 8; ++j) f[j]     = base[j];
  #pragma unroll
  for (int j = 0; j < 8; ++j) f[8 + j] = base[16 + j];
  return f;
}
// B: pre-swizzled fragment-order weights: per (ntile,ktile,lane) 16 contiguous
// halves (32B aligned) -> one v16h load, no cross-lane shuffling.
__device__ inline v16h load_frag_B_packed(const half_t* Bp, int KT, int nt, int kt){
  int lane = threadIdx.x & 31;
  const half_t* base = Bp + (((size_t)nt * KT + kt) * 32 + lane) * 16;
  v16h f;
  #pragma unroll
  for (int j = 0; j < 16; ++j) f[j] = base[j];
  return f;
}

// ---------------- kernel 0: weight packing f32 -> f16 fragment order --------
// P layout: [nt][kt][lane][16 halves], element e of lane holds
// k = kt*32 + (e<8?0:16) + (lane>>4)*8 + (e&7), n = nt*16 + (lane&15).
__device__ inline void pack_one(const float* W, half_t* P, int K, int N, int Ksrc, int i){
  int e    = i & 15;
  int lane = (i >> 4) & 31;
  int tile = i >> 9;                 // nt*KT + kt
  int KT   = K >> 5;
  int nt   = tile / KT, kt = tile - nt * KT;
  int n    = nt * 16 + (lane & 15);
  int hi   = lane >> 4;
  int k    = kt * 32 + ((e < 8) ? 0 : 16) + hi * 8 + (e & 7);
  float v  = (k < Ksrc) ? W[(size_t)k * N + n] : 0.0f;
  P[i] = (half_t)v;
}

__global__ __launch_bounds__(256)
void pack_weights_kernel(const float* Ww, const float* w0, const float* w1,
                         half_t* Wp, half_t* w0p, half_t* w1p){
  int i = blockIdx.x * blockDim.x + threadIdx.x;
  if (i < 64*64)   pack_one(Ww, Wp,  64,  64,  41, i);   // K padded 41->64
  if (i < 192*256) pack_one(w0, w0p, 192, 256, 192, i);
  if (i < 256*128) pack_one(w1, w1p, 256, 128, 256, i);
}

// ---------------- kernel 1: cosine top-K per batch (1 wave / batch) ---------
__global__ __launch_bounds__(32)
void topk_kernel(const int* __restrict__ x, const int* __restrict__ self_loc,
                 const float* __restrict__ emb, int* __restrict__ topk){
  int b = blockIdx.x, lane = threadIdx.x;
  __shared__ float sims[224];
  __shared__ float esh[32];
  __shared__ int   tk[16];
  int sl = self_loc[b];
  int self_cat = x[((size_t)b*Tlen + sl)*Ffeat + 7];
  float ev = emb[(size_t)self_cat*32 + lane];
  esh[lane] = ev;
  float n1 = sqrtf(wave_sum(ev*ev) + 1e-8f);
  __syncthreads();
  for (int t = lane; t < Tlen; t += 32){
    int cat = x[((size_t)b*Tlen + t)*Ffeat + 7];
    float dot = 0.f, nn = 0.f;
    #pragma unroll 8
    for (int d = 0; d < 32; ++d){ float e = emb[(size_t)cat*32 + d]; dot += esh[d]*e; nn += e*e; }
    float sim = dot / (n1 * sqrtf(nn + 1e-8f));
    sims[t] = (t < sl) ? sim : -2.0f;
  }
  __syncthreads();
  for (int k = 0; k < 15; ++k){
    float bv = -1e30f; int bi = 0x7fffffff;
    for (int t = lane; t < Tlen; t += 32){
      float v = sims[t];
      if (v > bv || (v == bv && t < bi)){ bv = v; bi = t; }
    }
    #pragma unroll
    for (int o = 16; o > 0; o >>= 1){
      float ov = __shfl_xor(bv, o, 32); int oi = __shfl_xor(bi, o, 32);
      if (ov > bv || (ov == bv && oi < bi)){ bv = ov; bi = oi; }
    }
    if (lane == 0){ tk[k] = bi; sims[bi] = -1e30f; }
    __syncthreads();
  }
  if (lane == 0){
    for (int i = 1; i < 15; ++i){           // ascending sort (matches jnp.sort)
      int key = tk[i], j = i - 1;
      while (j >= 0 && tk[j] > key){ tk[j+1] = tk[j]; --j; }
      tk[j+1] = key;
    }
    tk[15] = sl;
    for (int i = 0; i < 16; ++i) topk[b*Krows + i] = tk[i];
  }
}

// ---------------- kernel 2: PNN features -> f16 A matrix (1 wave / row) -----
__global__ __launch_bounds__(64)
void features_kernel(const int* __restrict__ x, const int* __restrict__ xc,
                     const int* __restrict__ topk,
                     const float* __restrict__ emb, const float* __restrict__ w1,
                     half_t* __restrict__ Afeat, int* __restrict__ maskArr){
  int b = blockIdx.x;
  int wave = threadIdx.x >> 5, lane = threadIdx.x & 31;
  for (int r = wave; r < ROWS_PER_B; r += 2){
    const int* row; float tcur, tprev; int isfirst;
    if (r < Krows){
      int src = topk[b*Krows + r];
      row  = x + ((size_t)b*Tlen + src)*Ffeat;
      tcur = (float)row[8];
      if (r == 0){ isfirst = 1; tprev = 0.f; }
      else { int ps = topk[b*Krows + r - 1];
             tprev = (float)x[((size_t)b*Tlen + ps)*Ffeat + 8]; isfirst = 0; }
    } else {
      int j = r - Krows;
      row  = xc + ((size_t)b*TClen + j)*Ffeat;
      tcur = (float)row[8];
      if (j == 0){ isfirst = 1; tprev = 0.f; }
      else { tprev = (float)xc[((size_t)b*TClen + j - 1)*Ffeat + 8]; isfirst = 0; }
    }
    float dt    = isfirst ? 0.f : (tcur - tprev);
    float tfeat = 1.0f / (dt + 1.0f);
    int ids[FKf];
    #pragma unroll
    for (int f = 0; f < FKf; ++f) ids[f] = row[f];
    float s = 0.f, sq = 0.f;                 // lane == embedding dim d
    #pragma unroll
    for (int f = 0; f < FKf; ++f){ float v = emb[(size_t)ids[f]*32 + lane]; s += v; sq += v*v; }
    float cross = 0.5f * (s*s - sq);
    size_t rowg = (size_t)b*ROWS_PER_B + r;
    half_t* A = Afeat + rowg*64;
    if (lane < FKf) A[lane]    = (half_t)w1[ids[lane]];
    A[FKf + lane]              = (half_t)cross;
    if (lane == 0)  A[40]      = (half_t)tfeat;
    if (lane < 23)  A[41+lane] = (half_t)0.f;     // zero K-pad 41..63
    if (lane == 0)  maskArr[rowg] = (ids[0] != VOCAB) ? 1 : 0;
  }
}

// ---------------- kernel 3: templated WMMA GEMM + bias (+ReLU) --------------
// 4 waves / block, each wave one 16x16 tile; grid = (M/64, N/16).
// Compile-time KP fully unrolls the k-loop -> straight-line WMMA chain.
template<int KP, int NN, bool RELU, bool OUTF32>
__global__ __launch_bounds__(128)
void wmma_gemm_kernel(const half_t* __restrict__ A, const half_t* __restrict__ Bp,
                      const float* __restrict__ bias, void* __restrict__ Cout){
  constexpr int KT = KP / 32;
  int wave = threadIdx.x >> 5;
  int lane = threadIdx.x & 31;
  int m0 = (blockIdx.x * 4 + wave) * 16;
  int nt = blockIdx.y;
  v8f acc = {};
  #pragma unroll
  for (int kt = 0; kt < KT; ++kt){
    v16h a = load_frag_A(A, KP, m0, kt * 32);
    v16h bf = load_frag_B_packed(Bp, KT, nt, kt);
    acc = __builtin_amdgcn_wmma_f32_16x16x32_f16(false, a, false, bf,
                                                 (short)0, acc, false, false);
  }
  int n  = nt * 16 + (lane & 15);
  int hi = lane >> 4;
  float bv = bias[n];
  #pragma unroll
  for (int r = 0; r < 8; ++r){
    float v = acc[r] + bv;
    if (RELU) v = fmaxf(v, 0.f);
    size_t idx = (size_t)(m0 + hi*8 + r) * NN + n;
    if (OUTF32) ((float*)Cout)[idx]  = v;
    else        ((half_t*)Cout)[idx] = (half_t)v;
  }
}

// ---------------- kernel 4: masked softmax attention + concat ---------------
// 2 waves / batch: wave0 = h1 (16 rows), wave1 = h2 (50 rows)
__global__ __launch_bounds__(64)
void attention_kernel(const float* __restrict__ h, const int* __restrict__ maskArr,
                      half_t* __restrict__ hcat){
  int b = blockIdx.x;
  int wave = threadIdx.x >> 5, lane = threadIdx.x & 31;
  __shared__ float alpha[2][64];
  __shared__ float qbuf[2][64];
  int rows = (wave == 0) ? Krows : TClen;
  int base = b*ROWS_PER_B + ((wave == 0) ? 0 : Krows);
  int qrow = base + rows - 1;
  qbuf[wave][lane]      = h[(size_t)qrow*Hdim + lane];
  qbuf[wave][lane + 32] = h[(size_t)qrow*Hdim + lane + 32];
  __syncthreads();
  float sc[2]; float mx = -3.0e38f;
  #pragma unroll
  for (int p = 0; p < 2; ++p){
    int j = lane + p*32;
    float s = -3.0e38f;
    if (j < rows){
      float acc = 0.f;
      const float* hr = h + (size_t)(base + j)*Hdim;
      #pragma unroll 8
      for (int d = 0; d < Hdim; ++d) acc += hr[d]*qbuf[wave][d];
      s = maskArr[base + j] ? acc : NEGV;
    }
    sc[p] = s; mx = fmaxf(mx, s);
  }
  mx = wave_max(mx);
  float sum = 0.f;
  #pragma unroll
  for (int p = 0; p < 2; ++p){
    int j = lane + p*32;
    float e = (j < rows) ? __expf(sc[p] - mx) : 0.f;
    alpha[wave][j] = e;
    sum += e;
  }
  sum = wave_sum(sum);
  __syncthreads();
  float inv = 1.0f / sum;
  #pragma unroll
  for (int p = 0; p < 2; ++p){
    int d = lane + p*32;
    float acc = 0.f;
    for (int j = 0; j < rows; ++j) acc += alpha[wave][j] * h[(size_t)(base + j)*Hdim + d];
    acc *= inv;
    hcat[(size_t)b*192 + ((wave == 0) ? 64 : 128) + d] = (half_t)acc;
    if (wave == 0) hcat[(size_t)b*192 + d] = (half_t)qbuf[0][d];   // x_self
  }
}

// ---------------- kernel 5: final 128-dim dot (1 wave / batch) --------------
__global__ __launch_bounds__(32)
void final_fc_kernel(const half_t* __restrict__ o2, const float* __restrict__ fcw,
                     const float* __restrict__ fcb, float* __restrict__ out){
  int b = blockIdx.x, lane = threadIdx.x;
  float acc = 0.f;
  #pragma unroll
  for (int d = lane; d < 128; d += 32) acc += (float)o2[(size_t)b*128 + d] * fcw[d];
  acc = wave_sum(acc);
  if (lane == 0) out[b] = acc + fcb[0];
}

// ---------------- host-side orchestration -----------------------------------
extern "C" void kernel_launch(void* const* d_in, const int* in_sizes, int n_in,
                              void* d_out, int out_size, void* d_ws, size_t ws_size,
                              hipStream_t stream){
  const int*   x    = (const int*)  d_in[0];
  const int*   xc   = (const int*)  d_in[1];
  const int*   sloc = (const int*)  d_in[2];
  const float* emb  = (const float*)d_in[3];
  const float* w1   = (const float*)d_in[4];
  const float* Ww   = (const float*)d_in[5];
  const float* Wb   = (const float*)d_in[6];
  const float* mw0  = (const float*)d_in[7];
  const float* mb0  = (const float*)d_in[8];
  const float* mw1  = (const float*)d_in[9];
  const float* mb1  = (const float*)d_in[10];
  const float* fcw  = (const float*)d_in[11];
  const float* fcb  = (const float*)d_in[12];
  float* out = (float*)d_out;

  char* ws = (char*)d_ws;
  size_t off = 0;
  auto take = [&](size_t bytes)->char*{
    char* p = ws + off; off = (off + bytes + 255) & ~(size_t)255; return p;
  };
  int*    topk  = (int*)   take((size_t)Bsz*Krows*4);
  half_t* Afeat = (half_t*)take((size_t)Mrows*64*2);
  int*    maskA = (int*)   take((size_t)Mrows*4);
  float*  hbuf  = (float*) take((size_t)Mrows*Hdim*4);
  half_t* hcat  = (half_t*)take((size_t)Bsz*192*2);
  half_t* o1    = (half_t*)take((size_t)Bsz*256*2);
  half_t* o2    = (half_t*)take((size_t)Bsz*128*2);
  half_t* Wp    = (half_t*)take((size_t)64*64*2);
  half_t* w0p   = (half_t*)take((size_t)192*256*2);
  half_t* w1p   = (half_t*)take((size_t)256*128*2);
  (void)ws_size; (void)in_sizes; (void)n_in; (void)out_size;

  // 0) pack weights to f16 in WMMA fragment order (zero-padded K for proj)
  pack_weights_kernel<<<192, 256, 0, stream>>>(Ww, mw0, mw1, Wp, w0p, w1p);
  // 1) cosine top-K retrieval
  topk_kernel<<<Bsz, 32, 0, stream>>>(x, sloc, emb, topk);
  // 2) PNN feature rows (f16, K padded to 64)
  features_kernel<<<Bsz, 64, 0, stream>>>(x, xc, topk, emb, w1, Afeat, maskA);
  // 3) projection GEMM: (67584,64) x (64,64) + W_b -> h (f32)
  wmma_gemm_kernel<64, 64, false, true><<<dim3(Mrows/64, Hdim/16), 128, 0, stream>>>(
      Afeat, Wp, Wb, hbuf);
  // 4) masked softmax attentions + concat -> hcat (f16, K=192)
  attention_kernel<<<Bsz, 64, 0, stream>>>(hbuf, maskA, hcat);
  // 5) MLP0: (1024,192)x(192,256)+b0, ReLU -> f16
  wmma_gemm_kernel<192, 256, true, false><<<dim3(Bsz/64, 256/16), 128, 0, stream>>>(
      hcat, w0p, mb0, o1);
  // 6) MLP1: (1024,256)x(256,128)+b1, ReLU -> f16
  wmma_gemm_kernel<256, 128, true, false><<<dim3(Bsz/64, 128/16), 128, 0, stream>>>(
      o1, w1p, mb1, o2);
  // 7) final FC: (1024,128)x(128,1)+fc_b
  final_fc_kernel<<<Bsz, 32, 0, stream>>>(o2, fcw, fcb, out);
}
